// JitGRU_Predictor_69801808494933
// MI455X (gfx1250) — compile-verified, matrix operand order checked by reading, not run on hardware
//
#include <hip/hip_runtime.h>
#include <hip/hip_bf16.h>

typedef __attribute__((ext_vector_type(16))) _Float16 v16h;
typedef __attribute__((ext_vector_type(8)))  float    v8f;

#define B_   256
#define T_   512
#define F_   8
#define H_   512
#define OUT_ 96

// fragment half-index -> K offset inside a 16x32 (or 32x16) tile
__device__ __forceinline__ int kOf(int e, int laneHi) {
    return (e & 7) + 8 * laneHi + 16 * (e >> 3);
}

__device__ __forceinline__ float sigm(float v) {
    return 1.f / (1.f + __expf(-v));
}
__device__ __forceinline__ float tanh_f(float v) {
    float e = __expf(2.f * v);
    return 1.f - 2.f / (e + 1.f);   // robust at +/- inf
}

// ---------------- pack kernels (run once, before the persistent kernel) ---------

// Whh packed: [gate(3)][jt(32)][kt(16)][lane(32)][e(16)] f16
__global__ void pack_whh(const float* __restrict__ whh, _Float16* __restrict__ dst) {
    int idx = blockIdx.x * blockDim.x + threadIdx.x;
    if (idx >= 3 * 32 * 16 * 32 * 16) return;
    int e  = idx & 15;
    int l  = (idx >> 4) & 31;
    int kt = (idx >> 9) & 15;
    int jt = (idx >> 13) & 31;
    int g  = idx >> 18;
    int row = g * 512 + jt * 16 + (l & 15);
    int col = kt * 32 + kOf(e, l >> 4);
    dst[idx] = (_Float16)whh[row * H_ + col];
}

// Extra K-tile B-fragments: [g(4)][jt(32)][lane(32)][e(16)] f16
// g=0: r  -> k<8: Wih_r, k==8: bih_r+bhh_r
// g=1: z  -> k<8: Wih_z, k==8: bih_z+bhh_z
// g=2: hn -> k==8: bhh_n (only)
// g=3: in -> k<8: Wih_n, k==8: bih_n
__global__ void pack_ext(const float* __restrict__ wih, const float* __restrict__ bih,
                         const float* __restrict__ bhh, _Float16* __restrict__ dst) {
    int idx = blockIdx.x * blockDim.x + threadIdx.x;
    if (idx >= 4 * 32 * 32 * 16) return;
    int e  = idx & 15;
    int l  = (idx >> 4) & 31;
    int jt = (idx >> 9) & 31;
    int g  = idx >> 14;
    int k  = kOf(e, l >> 4);
    int n  = l & 15;
    float v = 0.f;
    if (g == 0 || g == 1) {
        int row = g * 512 + jt * 16 + n;
        if (k < 8)       v = wih[row * F_ + k];
        else if (k == 8) v = bih[row] + bhh[row];
    } else if (g == 2) {
        int row = 1024 + jt * 16 + n;
        if (k == 8)      v = bhh[row];
    } else {
        int row = 1024 + jt * 16 + n;
        if (k < 8)       v = wih[row * F_ + k];
        else if (k == 8) v = bih[row];
    }
    dst[idx] = (_Float16)v;
}

// fc_w packed: [ot(6)][kt(16)][lane(32)][e(16)] f16
__global__ void pack_fcw(const float* __restrict__ fcw, _Float16* __restrict__ dst) {
    int idx = blockIdx.x * blockDim.x + threadIdx.x;
    if (idx >= 6 * 16 * 32 * 16) return;
    int e  = idx & 15;
    int l  = (idx >> 4) & 31;
    int kt = (idx >> 9) & 15;
    int ot = idx >> 13;
    int row = ot * 16 + (l & 15);
    int col = kt * 32 + kOf(e, l >> 4);
    dst[idx] = (_Float16)fcw[row * H_ + col];
}

// ---------------- persistent GRU kernel: one WG per 16 batch rows ---------------
// 512 threads = 16 waves; wave w owns column tiles jt = 2w, 2w+1.

__global__ __launch_bounds__(512) void gru_persistent(
    const float*    __restrict__ x,     // [B, T, F]
    const _Float16* __restrict__ whhP,  // [3][32][16][32][16]
    const _Float16* __restrict__ extP,  // [4][32][32][16]
    const _Float16* __restrict__ fcwP,  // [6][16][32][16]
    const float*    __restrict__ fcb,   // [96]
    float*          __restrict__ out)   // [B, 96]
{
    // 17 A-tiles (16 for h, tile 16 = [x_t | 1 | 0...]), each 32 lanes x 16 halves
    __shared__ _Float16 aFrag[17 * 512] __attribute__((aligned(32)));

    const int tid    = threadIdx.x;
    const int lane   = tid & 31;
    const int wave   = tid >> 5;
    const int laneHi = lane >> 4;
    const int nIdx   = lane & 15;
    const int B0     = blockIdx.x * 16;

    // ---- init: zero h frags, build static part of tile 16 (k==8 -> 1.0) ----
    for (int i = tid; i < 16 * 512; i += 512) aFrag[i] = (_Float16)0.f;
    if (tid < 512) {
        int l = tid >> 4, e = tid & 15;
        int k = kOf(e, l >> 4);
        aFrag[16 * 512 + tid] = (_Float16)((k == 8) ? 1.f : 0.f);
    }
    // x_0 into tile 16 (lanes 0-15, e=0..7 <=> k=0..7)
    if (tid < 16) {
        const float* xr = x + ((size_t)(B0 + tid) * T_ + 0) * F_;
        #pragma unroll
        for (int f = 0; f < 8; ++f)
            aFrag[16 * 512 + tid * 16 + f] = (_Float16)xr[f];
    }

    v8f hreg[2];   // f32 master copy of h for this wave's 2 column tiles
    hreg[0] = (v8f){};
    hreg[1] = (v8f){};

    for (int t = 0; t < T_; ++t) {
        __syncthreads();   // aFrag writes visible to all waves

        #pragma unroll
        for (int u = 0; u < 2; ++u) {
            const int jt = wave * 2 + u;
            v8f ar = (v8f){}, az = (v8f){}, ahn = (v8f){}, ain = (v8f){};

            const _Float16* br0 = &whhP[(((0 * 32 + jt) * 16) * 32 + lane) * 16];
            const _Float16* bz0 = &whhP[(((1 * 32 + jt) * 16) * 32 + lane) * 16];
            const _Float16* bh0 = &whhP[(((2 * 32 + jt) * 16) * 32 + lane) * 16];

            #pragma unroll 4
            for (int kt = 0; kt < 16; ++kt) {
                v16h a  = *(const v16h*)&aFrag[(kt * 32 + lane) * 16];
                v16h br = *(const v16h*)&br0[kt * 512];
                v16h bz = *(const v16h*)&bz0[kt * 512];
                v16h bh = *(const v16h*)&bh0[kt * 512];
                ar  = __builtin_amdgcn_wmma_f32_16x16x32_f16(false, a, false, br, (short)0, ar,  false, false);
                az  = __builtin_amdgcn_wmma_f32_16x16x32_f16(false, a, false, bz, (short)0, az,  false, false);
                ahn = __builtin_amdgcn_wmma_f32_16x16x32_f16(false, a, false, bh, (short)0, ahn, false, false);
            }
            {   // extra K-tile: x-projection + biases
                v16h a  = *(const v16h*)&aFrag[(16 * 32 + lane) * 16];
                v16h br = *(const v16h*)&extP[((0 * 32 + jt) * 32 + lane) * 16];
                v16h bz = *(const v16h*)&extP[((1 * 32 + jt) * 32 + lane) * 16];
                v16h bh = *(const v16h*)&extP[((2 * 32 + jt) * 32 + lane) * 16];
                v16h bi = *(const v16h*)&extP[((3 * 32 + jt) * 32 + lane) * 16];
                ar  = __builtin_amdgcn_wmma_f32_16x16x32_f16(false, a, false, br, (short)0, ar,  false, false);
                az  = __builtin_amdgcn_wmma_f32_16x16x32_f16(false, a, false, bz, (short)0, az,  false, false);
                ahn = __builtin_amdgcn_wmma_f32_16x16x32_f16(false, a, false, bh, (short)0, ahn, false, false);
                ain = __builtin_amdgcn_wmma_f32_16x16x32_f16(false, a, false, bi, (short)0, (v8f){}, false, false);
            }
            // elementwise gate math (VALU, co-executes with XDL WMMA)
            #pragma unroll
            for (int v2 = 0; v2 < 8; ++v2) {
                float r  = sigm(ar[v2]);
                float z  = sigm(az[v2]);
                float nn = tanh_f(ain[v2] + r * ahn[v2]);
                hreg[u][v2] = nn + z * (hreg[u][v2] - nn);
            }
        }

        __syncthreads();   // everyone finished reading aFrag

        // scatter h_new (f16) into A-fragment layout for the next step
        #pragma unroll
        for (int u = 0; u < 2; ++u) {
            const int jt   = wave * 2 + u;
            const int tIdx = jt >> 1;
            const int kk   = nIdx + 16 * (jt & 1);          // K position inside tile
            const int e    = (kk & 7) | ((kk >> 4) << 3);
            const int lh   = (kk >> 3) & 1;
            #pragma unroll
            for (int v2 = 0; v2 < 8; ++v2) {
                int m  = v2 + 8 * laneHi;                   // row (batch-local)
                int lp = m + 16 * lh;                       // target lane
                aFrag[tIdx * 512 + lp * 16 + e] = (_Float16)hreg[u][v2];
            }
        }
        // x_{t+1} into tile 16
        if (t + 1 < T_ && tid < 16) {
            const float* xr = x + ((size_t)(B0 + tid) * T_ + (t + 1)) * F_;
            #pragma unroll
            for (int f = 0; f < 8; ++f)
                aFrag[16 * 512 + tid * 16 + f] = (_Float16)xr[f];
        }
    }

    __syncthreads();   // h_last frags visible

    // ---- final FC: out[b, o] = h_last @ fc_w^T + fc_b  (waves 0..5, 16 cols each)
    if (wave < 6) {
        const int ot = wave;
        float bv = fcb[ot * 16 + nIdx];
        v8f acc;
        #pragma unroll
        for (int v2 = 0; v2 < 8; ++v2) acc[v2] = bv;
        #pragma unroll 4
        for (int kt = 0; kt < 16; ++kt) {
            v16h a = *(const v16h*)&aFrag[(kt * 32 + lane) * 16];
            v16h b = *(const v16h*)&fcwP[((ot * 16 + kt) * 32 + lane) * 16];
            acc = __builtin_amdgcn_wmma_f32_16x16x32_f16(false, a, false, b, (short)0, acc, false, false);
        }
        #pragma unroll
        for (int v2 = 0; v2 < 8; ++v2) {
            int m = v2 + 8 * laneHi;
            out[(size_t)(B0 + m) * OUT_ + ot * 16 + nIdx] = acc[v2];
        }
    }
}

// ---------------- host side --------------------------------------------------

extern "C" void kernel_launch(void* const* d_in, const int* in_sizes, int n_in,
                              void* d_out, int out_size, void* d_ws, size_t ws_size,
                              hipStream_t stream) {
    const float* x   = (const float*)d_in[0];
    const float* wih = (const float*)d_in[1];
    const float* whh = (const float*)d_in[2];
    const float* bih = (const float*)d_in[3];
    const float* bhh = (const float*)d_in[4];
    const float* fcw = (const float*)d_in[5];
    const float* fcb = (const float*)d_in[6];
    float* out = (float*)d_out;

    _Float16* wsWhh = (_Float16*)d_ws;                 // 786432 halves
    _Float16* wsExt = wsWhh + 3 * 32 * 16 * 32 * 16;   // 65536 halves
    _Float16* wsFcw = wsExt + 4 * 32 * 32 * 16;        // 49152 halves

    pack_whh<<<(3 * 32 * 16 * 32 * 16 + 255) / 256, 256, 0, stream>>>(whh, wsWhh);
    pack_ext<<<(4 * 32 * 32 * 16 + 255) / 256, 256, 0, stream>>>(wih, bih, bhh, wsExt);
    pack_fcw<<<(6 * 16 * 32 * 16 + 255) / 256, 256, 0, stream>>>(fcw, wsFcw);

    gru_persistent<<<B_ / 16, 512, 0, stream>>>(x, wsWhh, wsExt, wsFcw, fcb, out);
}
